// LTRSubNetwork_36094905155880
// MI455X (gfx1250) — matrix operand order repaired and checked by reading, not compiled, not run on hardware
//
#include <hip/hip_runtime.h>
#include <hip/hip_bf16.h>

// ---------------------------------------------------------------------------
// Problem sizes (compile-time constants from the reference)
// ---------------------------------------------------------------------------
#define SENT_NUM 2048
#define ROI_NUM  4096
#define HEAD_DIM 2048
#define FEAT_DIM 1024

typedef __attribute__((ext_vector_type(16))) __bf16 v16bf;
typedef __attribute__((ext_vector_type(8)))  float  v8f;

union Frag {
    v16bf v;
    uint4 q[2];
};

__device__ __forceinline__ unsigned short f2bf_rne(float x) {
    unsigned u = __float_as_uint(x);
    unsigned r = u + 0x7FFFu + ((u >> 16) & 1u);   // round-to-nearest-even
    return (unsigned short)(r >> 16);
}

// ---------------------------------------------------------------------------
// Elementwise f32 -> bf16 conversion (vectorized x4)
// ---------------------------------------------------------------------------
__global__ __launch_bounds__(256) void cvt_f32_to_bf16(
    const float* __restrict__ in, unsigned short* __restrict__ out) {
    const size_t i = ((size_t)blockIdx.x * blockDim.x + threadIdx.x) * 4;
    float4 f = *(const float4*)(in + i);
    ushort4 o;
    o.x = f2bf_rne(f.x);
    o.y = f2bf_rne(f.y);
    o.z = f2bf_rne(f.z);
    o.w = f2bf_rne(f.w);
    *(ushort4*)(out + i) = o;
}

// sent_bf[s,d] = bf16(sent_feat[s,d] * rank_w[d])  (folds rank vector into GEMM2 A)
__global__ __launch_bounds__(256) void scale_cvt_bf16(
    const float* __restrict__ sent, const float* __restrict__ rankw,
    unsigned short* __restrict__ out) {
    const size_t i = ((size_t)blockIdx.x * blockDim.x + threadIdx.x) * 4;
    const int d = (int)(i & (FEAT_DIM - 1));
    float4 s = *(const float4*)(sent + i);
    float4 w = *(const float4*)(rankw + d);
    ushort4 o;
    o.x = f2bf_rne(s.x * w.x);
    o.y = f2bf_rne(s.y * w.y);
    o.z = f2bf_rne(s.z * w.z);
    o.w = f2bf_rne(s.w * w.w);
    *(ushort4*)(out + i) = o;
}

// ---------------------------------------------------------------------------
// NT GEMM: C[M,N] = A[M,K] * B[N,K]^T, A/B row-major bf16, f32 accumulate.
//   MODE 0: C = bf16(relu(acc + bias[n]))   (bf16 output, row-major [M,N])
//   MODE 1: C = acc                          (f32 output, row-major [M,N])
// Block = 128 threads = 4 waves. Block tile 128x128, wave tile 64x64
// = 4x4 grid of 16x16 WMMA tiles, K-step 32 (v_wmma_f32_16x16x32_bf16).
//
// Fragment layouts (CDNA5 ISA 7.12.2), lane = l, half = l>>4, l16 = l&15:
//   A frag: row m = l16; ks = {kb + half*8 + 0..7} U {kb + 16 + half*8 + 0..7}
//           -> two contiguous 16B loads from row-major A.
//   B frag: col n = l16; ks = {kb + half*8 + 0..7} U {kb + 16 + half*8 + 0..7}
//           -> two contiguous 16B loads from row-major B (B stored [N,K]).
//   C/D:    elem v of v8f is (m = v + 8*half, n = l16).
//
// Prefetch: unconditional (speculative prefetch past the end of the buffer is
// silently dropped per ISA §10.5) — keeps the inner loop branch-free.
// ---------------------------------------------------------------------------
template <int MODE>
__global__ __launch_bounds__(128) void gemm_nt_wmma_bf16(
    const unsigned short* __restrict__ A,
    const unsigned short* __restrict__ B,
    const float* __restrict__ bias,
    void* __restrict__ C,
    int M, int N, int K) {
    const int lane = threadIdx.x & 31;
    const int wave = threadIdx.x >> 5;
    const int half = lane >> 4;
    const int l16  = lane & 15;

    const int wm = blockIdx.y * 128 + (wave >> 1) * 64;  // wave row base
    const int wn = blockIdx.x * 128 + (wave & 1) * 64;   // wave col base

    v8f acc[4][4];
#pragma unroll
    for (int mt = 0; mt < 4; ++mt)
#pragma unroll
        for (int nt = 0; nt < 4; ++nt)
            acc[mt][nt] = (v8f)0.0f;

    for (int kb = 0; kb < K; kb += 32) {
        Frag fa[4], fb[4];
#pragma unroll
        for (int t = 0; t < 4; ++t) {
            const unsigned short* pa =
                A + (size_t)(wm + t * 16 + l16) * K + kb + half * 8;
            fa[t].q[0] = *(const uint4*)(pa);
            fa[t].q[1] = *(const uint4*)(pa + 16);
            __builtin_prefetch(pa + 32, 0, 3);  // next K block (speculative)
            const unsigned short* pb =
                B + (size_t)(wn + t * 16 + l16) * K + kb + half * 8;
            fb[t].q[0] = *(const uint4*)(pb);
            fb[t].q[1] = *(const uint4*)(pb + 16);
            __builtin_prefetch(pb + 32, 0, 3);
        }
#pragma unroll
        for (int mt = 0; mt < 4; ++mt)
#pragma unroll
            for (int nt = 0; nt < 4; ++nt)
                acc[mt][nt] = __builtin_amdgcn_wmma_f32_16x16x32_bf16(
                    /*neg_a=*/false, fa[mt].v, /*neg_b=*/false, fb[nt].v,
                    /*c_mod=*/(short)0, acc[mt][nt],
                    /*reuse_a=*/false, /*reuse_b=*/false);
    }

    if (MODE == 0) {
        unsigned short* Cs = (unsigned short*)C;
#pragma unroll
        for (int mt = 0; mt < 4; ++mt)
#pragma unroll
            for (int nt = 0; nt < 4; ++nt) {
                const int n = wn + nt * 16 + l16;
                const float b = bias[n];
#pragma unroll
                for (int v = 0; v < 8; ++v) {
                    const int m = wm + mt * 16 + half * 8 + v;
                    float x = acc[mt][nt][v] + b;
                    x = x > 0.0f ? x : 0.0f;
                    Cs[(size_t)m * N + n] = f2bf_rne(x);
                }
            }
    } else {
        float* Cf = (float*)C;
#pragma unroll
        for (int mt = 0; mt < 4; ++mt)
#pragma unroll
            for (int nt = 0; nt < 4; ++nt) {
                const int n = wn + nt * 16 + l16;
#pragma unroll
                for (int v = 0; v < 8; ++v) {
                    const int m = wm + mt * 16 + half * 8 + v;
                    Cf[(size_t)m * N + n] = acc[mt][nt][v];
                }
            }
    }
}

// ---------------------------------------------------------------------------
// Launch: inputs in setup_inputs() order:
//   d_in[0] sent_feat [2048,1024] f32
//   d_in[1] head_feat [4096,2048] f32
//   d_in[2] head_w    [1024,2048] f32
//   d_in[3] head_b    [1024]      f32
//   d_in[4] rank_w    [1,1024]    f32
// d_out: rank_score [2048,4096] f32
// ---------------------------------------------------------------------------
extern "C" void kernel_launch(void* const* d_in, const int* in_sizes, int n_in,
                              void* d_out, int out_size, void* d_ws, size_t ws_size,
                              hipStream_t stream) {
    const float* sent_feat = (const float*)d_in[0];
    const float* head_feat = (const float*)d_in[1];
    const float* head_w    = (const float*)d_in[2];
    const float* head_b    = (const float*)d_in[3];
    const float* rank_w    = (const float*)d_in[4];

    // Workspace layout (bf16 elements), total 32 MB:
    unsigned short* headf_bf  = (unsigned short*)d_ws;                       // R*H  (16 MB)
    unsigned short* headw_bf  = headf_bf + (size_t)ROI_NUM * HEAD_DIM;       // D*H  ( 4 MB)
    unsigned short* sent_bf   = headw_bf + (size_t)FEAT_DIM * HEAD_DIM;      // S*D  ( 4 MB)
    unsigned short* mapped_bf = sent_bf  + (size_t)SENT_NUM * FEAT_DIM;      // R*D  ( 8 MB)

    // 1) Convert inputs to bf16 (bandwidth-bound, x4 vectorized)
    {
        const size_t n1 = (size_t)ROI_NUM * HEAD_DIM;   // 8M
        cvt_f32_to_bf16<<<(unsigned)(n1 / 4 / 256), 256, 0, stream>>>(head_feat, headf_bf);
        const size_t n2 = (size_t)FEAT_DIM * HEAD_DIM;  // 2M
        cvt_f32_to_bf16<<<(unsigned)(n2 / 4 / 256), 256, 0, stream>>>(head_w, headw_bf);
        const size_t n3 = (size_t)SENT_NUM * FEAT_DIM;  // 2M
        scale_cvt_bf16<<<(unsigned)(n3 / 4 / 256), 256, 0, stream>>>(sent_feat, rank_w, sent_bf);
    }

    // 2) GEMM1: mapped[R,D] = relu(head_feat[R,H] @ head_w[D,H]^T + head_b), bf16 out
    {
        dim3 grid(FEAT_DIM / 128, ROI_NUM / 128);  // (8, 32)
        gemm_nt_wmma_bf16<0><<<grid, 128, 0, stream>>>(
            headf_bf, headw_bf, head_b, mapped_bf, ROI_NUM, FEAT_DIM, HEAD_DIM);
    }

    // 3) GEMM2: rank_score[S,R] = sent_bf[S,D] @ mapped[R,D]^T, f32 out
    {
        dim3 grid(ROI_NUM / 128, SENT_NUM / 128);  // (32, 16)
        gemm_nt_wmma_bf16<1><<<grid, 128, 0, stream>>>(
            sent_bf, mapped_bf, nullptr, d_out, SENT_NUM, ROI_NUM, FEAT_DIM);
    }
}
